// NT_Xent_41626823033666
// MI455X (gfx1250) — matrix operand order repaired and checked by reading, not compiled, not run on hardware
//
#include <hip/hip_runtime.h>
#include <hip/hip_bf16.h>
#include <math.h>

typedef __attribute__((ext_vector_type(16))) _Float16 v16h;
typedef __attribute__((ext_vector_type(8)))  _Float16 v8h;
typedef __attribute__((ext_vector_type(8)))  float    v8f;
typedef __attribute__((ext_vector_type(4)))  float    v4f;

#define NROWS   8192        // N = 2B
#define BHALF   4096        // B
#define DDIM    128
#define EPS_NORM 1e-8f
#define TWO_OVER_LN2 2.8853900817779268f   // exp(2x) = exp2(x * 2/ln2)

#define COLS_PER_STAGE 64
#define NSTAGES (NROWS / COLS_PER_STAGE)   // 128

// ---- gfx1250 async global->LDS path (probe via __has_builtin) -------------
#if defined(__has_builtin)
#if __has_builtin(__builtin_amdgcn_global_load_async_to_lds_b128)
#define HAVE_ASYNC_LDS 1
#endif
#endif
#ifndef HAVE_ASYNC_LDS
#define HAVE_ASYNC_LDS 0
#endif

#if HAVE_ASYNC_LDS
typedef int v4i_vs __attribute__((vector_size(16)));
typedef v4i_vs __attribute__((address_space(1)))* gptr_v4i;  // global src
typedef v4i_vs __attribute__((address_space(3)))* lptr_v4i;  // LDS dst
#endif

// Raw v_exp_f32 (arguments are bounded, skip libm's subnormal fixup).
static __device__ __forceinline__ float fast_exp2(float x) {
#if defined(__has_builtin)
#if __has_builtin(__builtin_amdgcn_exp2f)
  return __builtin_amdgcn_exp2f(x);
#else
  return __expf(x * 0.6931471805599453f);
#endif
#else
  return __expf(x * 0.6931471805599453f);
#endif
}

static __device__ __forceinline__ void async_wait0() {
#if HAVE_ASYNC_LDS
#if __has_builtin(__builtin_amdgcn_s_wait_asynccnt)
  __builtin_amdgcn_s_wait_asynccnt(0);
#else
  asm volatile("s_wait_asynccnt 0x0" ::: "memory");
#endif
#endif
}

// Copy one 64-row x 128-half column panel (16 KB) into LDS: 128 threads,
// 8 x 16-byte chunks each, via CDNA5 async-to-LDS (ASYNCcnt) when available.
static __device__ __forceinline__ void stage_fill(const _Float16* __restrict__ zn,
                                                  int colPanel,
                                                  _Float16* __restrict__ buf,
                                                  int tid) {
  const char* gbase = (const char*)(zn + (size_t)colPanel * DDIM);
  char* lbase = (char*)buf;
#pragma unroll
  for (int c = 0; c < 8; ++c) {
    const int byteOff = (c * 128 + tid) * 16;
#if HAVE_ASYNC_LDS
    __builtin_amdgcn_global_load_async_to_lds_b128(
        (gptr_v4i)(gbase + byteOff), (lptr_v4i)(lbase + byteOff),
        /*imm offset=*/0, /*cpol=*/0);
#else
    *(v8h*)(lbase + byteOff) = *(const v8h*)(gbase + byteOff);
#endif
  }
}

// Load the 4 K-slice B fragments (32x16 f16 each) for one 16-column tile.
// B layout (ISA 7.12.2): lane l holds col N=l%16; 8-half chunks at
// K = 32s + 8g and K = 32s + 16 + 8g. Emits 8x ds_load_b128 with no intervening WMMA.
static __device__ __forceinline__ void load_bfrag(v16h b[4],
                                                  const _Float16* __restrict__ colBase,
                                                  int g) {
  const v8h* colp = (const v8h*)colBase;
#pragma unroll
  for (int s = 0; s < 4; ++s) {
    v8h lo = colp[4 * s + g];          // K = 32s + 8g + (0..7)
    v8h hi = colp[4 * s + 2 + g];      // K = 32s + 16 + 8g + (0..7)
#pragma unroll
    for (int i = 0; i < 8; ++i) { b[s][i] = lo[i]; b[s][i + 8] = hi[i]; }
  }
}

// ---------------------------------------------------------------------------
// Kernel 1: L2-normalize rows of [z_i ; z_j] into f16 zn[N][128].
// ---------------------------------------------------------------------------
__global__ void nt_xent_normalize(const float* __restrict__ zi,
                                  const float* __restrict__ zj,
                                  _Float16* __restrict__ zn) {
  const int wave = (blockIdx.x * blockDim.x + threadIdx.x) >> 5;
  const int lane = threadIdx.x & 31;
  if (wave >= NROWS) return;
  const float* src = (wave < BHALF) ? (zi + (size_t)wave * DDIM)
                                    : (zj + (size_t)(wave - BHALF) * DDIM);
  v4f v = *(const v4f*)(src + lane * 4);
  float ss = v.x * v.x + v.y * v.y + v.z * v.z + v.w * v.w;
#pragma unroll
  for (int m = 16; m >= 1; m >>= 1) ss += __shfl_xor(ss, m, 32);
  const float inv = 1.0f / fmaxf(sqrtf(ss), EPS_NORM);
  _Float16* dst = zn + (size_t)wave * DDIM + lane * 4;
  dst[0] = (_Float16)(v.x * inv);
  dst[1] = (_Float16)(v.y * inv);
  dst[2] = (_Float16)(v.z * inv);
  dst[3] = (_Float16)(v.w * inv);
}

// ---------------------------------------------------------------------------
// Kernel 2: fused sim-GEMM + softmax denominator + positive capture.
// 128 threads = 4 waves per block; block owns 64 rows (wave w -> 16-row strip).
// Column panel staged to LDS (double-buffered via async-to-LDS) and shared by
// all 4 waves. B fragments register-double-buffered across the 4 sub-tiles so
// each 4-WMMA chain issues back-to-back behind already-issued ds_loads.
// ---------------------------------------------------------------------------
__global__ void __launch_bounds__(128)
nt_xent_sim(const _Float16* __restrict__ zn, float* __restrict__ rowLoss) {
  __shared__ __align__(16) _Float16 lds[2][COLS_PER_STAGE * DDIM];  // 2 x 16KB

  const int tid  = threadIdx.x;
  const int lane = tid & 31;
  const int wv   = tid >> 5;
  const int rowBase = (blockIdx.x * 4 + wv) * 16;
  const int g  = lane >> 4;
  const int lm = lane & 15;
  // Wave-uniform tile ids -> SGPRs so tile classification is scalar-branched.
  const int selfTile = __builtin_amdgcn_readfirstlane(rowBase >> 4);
  const int partnerTile = __builtin_amdgcn_readfirstlane(
      (rowBase >> 4) + ((rowBase < BHALF) ? (BHALF / 16) : -(BHALF / 16)));

  // --- A fragments: rows rowBase..rowBase+15, K = 0..127, kept in VGPRs ---
  // A layout (ISA 7.12.2): lane l holds row M=l%16; 8-half chunks at
  // K = 32s + 8g and K = 32s + 16 + 8g.
  v16h a[4];
  {
    const v8h* rowp = (const v8h*)(zn + (size_t)(rowBase + lm) * DDIM);
#pragma unroll
    for (int s = 0; s < 4; ++s) {
      v8h lo = rowp[4 * s + g];
      v8h hi = rowp[4 * s + 2 + g];
#pragma unroll
      for (int i = 0; i < 8; ++i) { a[s][i] = lo[i]; a[s][i + 8] = hi[i]; }
    }
  }

  float rs[8], ps[8];
#pragma unroll
  for (int i = 0; i < 8; ++i) { rs[i] = 0.0f; ps[i] = 0.0f; }

  // --- prologue: fill stage 0 ---
  stage_fill(zn, 0, lds[0], tid);
  async_wait0();
  __syncthreads();

  int buf = 0;
  for (int st = 0; st < NSTAGES; ++st) {
    // async-prefetch next panel into the other buffer during compute
    if (st + 1 < NSTAGES)
      stage_fill(zn, (st + 1) * COLS_PER_STAGE, lds[buf ^ 1], tid);

    // register double-buffer of B fragments across the 4 column sub-tiles
    v16h bcur[4], bnxt[4];
    load_bfrag(bcur, &lds[buf][(size_t)lm * DDIM], g);

#pragma unroll
    for (int sub = 0; sub < 4; ++sub) {
      if (sub < 3)
        load_bfrag(bnxt, &lds[buf][(size_t)((sub + 1) * 16 + lm) * DDIM], g);

      v8f c = {};
#pragma unroll
      for (int s = 0; s < 4; ++s)
        c = __builtin_amdgcn_wmma_f32_16x16x32_f16(
                false, a[s], false, bcur[s], (short)0, c, false, false);

      // Epilogue: exp(sim/T) with T = 0.5; exp(2c) = exp2(c * 2/ln2),
      // via raw v_exp_f32 (arguments bounded by |c|<=1, no fixup needed).
      const int ct = st * 4 + sub;
      if (ct == selfTile) {
#pragma unroll
        for (int v = 0; v < 8; ++v) {
          const float e = fast_exp2(c[v] * TWO_OVER_LN2);
          rs[v] += (lm == v + 8 * g) ? 0.0f : e;       // exclude diagonal
        }
      } else if (ct == partnerTile) {
#pragma unroll
        for (int v = 0; v < 8; ++v) {
          const float e = fast_exp2(c[v] * TWO_OVER_LN2);
          rs[v] += e;
          if (lm == v + 8 * g) ps[v] = 2.0f * c[v];    // positive pair sim/T
        }
      } else {
#pragma unroll
        for (int v = 0; v < 8; ++v) rs[v] += fast_exp2(c[v] * TWO_OVER_LN2);
      }

#pragma unroll
      for (int s = 0; s < 4; ++s) bcur[s] = bnxt[s];   // renamed away (unrolled)
    }

    async_wait0();       // next panel resident (this wave's share)
    __syncthreads();     // all waves done reading buf & all fills visible
    buf ^= 1;
  }

  // Reduce across the 16 lanes sharing each row (keep lane halves apart).
#pragma unroll
  for (int mask = 8; mask >= 1; mask >>= 1) {
#pragma unroll
    for (int v = 0; v < 8; ++v) {
      rs[v] += __shfl_xor(rs[v], mask, 16);
      ps[v] += __shfl_xor(ps[v], mask, 16);   // only one lane held nonzero
    }
  }
  if (lm == 0) {
#pragma unroll
    for (int v = 0; v < 8; ++v)
      rowLoss[rowBase + v + 8 * g] = __logf(rs[v]) - ps[v];
  }
}

// ---------------------------------------------------------------------------
// Kernel 3: deterministic fixed-order reduction -> mean -> d_out[0].
// ---------------------------------------------------------------------------
__global__ void nt_xent_reduce(const float* __restrict__ rowLoss,
                               float* __restrict__ out) {
  __shared__ float sm[256];
  float s = 0.0f;
  for (int i = threadIdx.x; i < NROWS; i += 256) s += rowLoss[i];
  sm[threadIdx.x] = s;
  __syncthreads();
  for (int w = 128; w >= 1; w >>= 1) {
    if ((int)threadIdx.x < w) sm[threadIdx.x] += sm[threadIdx.x + w];
    __syncthreads();
  }
  if (threadIdx.x == 0) out[0] = sm[0] * (1.0f / (float)NROWS);
}

// ---------------------------------------------------------------------------
extern "C" void kernel_launch(void* const* d_in, const int* in_sizes, int n_in,
                              void* d_out, int out_size, void* d_ws, size_t ws_size,
                              hipStream_t stream) {
  const float* zi = (const float*)d_in[0];
  const float* zj = (const float*)d_in[1];

  _Float16* zn      = (_Float16*)d_ws;                                   // 2 MB
  float*    rowLoss = (float*)((char*)d_ws +
                               (size_t)NROWS * DDIM * sizeof(_Float16)); // 32 KB

  nt_xent_normalize<<<NROWS / 8, 256, 0, stream>>>(zi, zj, zn);
  nt_xent_sim<<<NROWS / 64, 128, 0, stream>>>(zn, rowLoss);   // 128 blocks x 4 waves
  nt_xent_reduce<<<1, 256, 0, stream>>>(rowLoss, (float*)d_out);
}
